// SceneGraph_89790586290370
// MI455X (gfx1250) — matrix operand order repaired and checked by reading, not compiled, not run on hardware
//
#include <hip/hip_runtime.h>
#include <cstdint>
#include <cstddef>

// ---------------- types ----------------
typedef __attribute__((ext_vector_type(16))) __bf16 v16bf;
typedef __attribute__((ext_vector_type(8)))  __bf16 v8bf;
typedef __attribute__((ext_vector_type(8)))  float  v8f;
typedef __attribute__((ext_vector_type(4)))  float  v4f;

#define NNODE 192
#define DDIM  1024
#define EDIM  256
#define NEDGE (192*191)   // 36672 = 573*64

__device__ __forceinline__ float gelu_f(float v) {
    return 0.5f * v * (1.0f + erff(v * 0.70710678118654752440f));
}

// ---- WMMA fragment loaders ----
// A (16x32 MxK) row pointer: lane m = lane%16, halves hold K {0..7,16..23}/{8..15,24..31}
__device__ __forceinline__ v16bf afrag_from(const __bf16* rowp, int k0, int half) {
    v8bf a0 = *(const v8bf*)(rowp + k0 + half * 8);
    v8bf a1 = *(const v8bf*)(rowp + k0 + 16 + half * 8);
    return __builtin_shufflevector(a0, a1, 0,1,2,3,4,5,6,7,8,9,10,11,12,13,14,15);
}
// B (32x16 KxN) staged [n][k]: lane n = lane%16, halves hold K {0..7,16..23}/{8..15,24..31}
__device__ __forceinline__ v16bf bfrag_from(const __bf16* rowp, int k0, int half) {
    v8bf b0 = *(const v8bf*)(rowp + k0 + half * 8);
    v8bf b1 = *(const v8bf*)(rowp + k0 + 16 + half * 8);
    return __builtin_shufflevector(b0, b1, 0,1,2,3,4,5,6,7,8,9,10,11,12,13,14,15);
}

// ============== pack: fp32 B[K][N] -> bf16 Bp[N][K] (tiled transpose) ==============
__global__ __launch_bounds__(256) void pack_b(
    const float* __restrict__ B, int ldb, __bf16* __restrict__ Bp, int K)
{
    __shared__ float tile[64][68];
    const int kt = blockIdx.y * 64, nt = blockIdx.x * 64;
    const int tid = threadIdx.x;
    {   // load 64(k) x 64(n), coalesced in n
        int r = tid >> 2, c4 = (tid & 3) * 16;
        const float* src = B + (size_t)(kt + r) * ldb + nt + c4;
        #pragma unroll
        for (int i = 0; i < 4; ++i)
            *(v4f*)&tile[r][c4 + 4 * i] = *(const v4f*)(src + 4 * i);
    }
    __syncthreads();
    {   // store 64(n) x 64(k), coalesced in k, converted to bf16
        int rn = tid >> 2, kk = (tid & 3) * 16;
        __bf16* dst = Bp + (size_t)(nt + rn) * K + kt + kk;
        v8bf o0, o1;
        #pragma unroll
        for (int i = 0; i < 8; ++i) {
            o0[i] = (__bf16)tile[kk + i][rn];
            o1[i] = (__bf16)tile[kk + 8 + i][rn];
        }
        *(v8bf*)dst = o0;
        *(v8bf*)(dst + 8) = o1;
    }
}

// ============== generic tiled WMMA GEMM (small GEMMs) ==============
// C[64x64 tile] = A[M,K](fp32, lda) @ Bp[N,K](bf16 packed) + bias
// Requires M%64==0, N%64==0, K%32==0.
__global__ __launch_bounds__(256) void gemm_wmma(
    const float* __restrict__ A, int lda,
    const __bf16* __restrict__ Bp,
    const float* __restrict__ bias,
    float* __restrict__ C, int ldc, int K)
{
    __shared__ alignas(128) __bf16 As[64][32];
    __shared__ alignas(128) __bf16 Bs[64][32];

    const int tid  = threadIdx.x;
    const int lane = tid & 31, wave = tid >> 5;
    const int wm = wave & 3, wn = wave >> 2;     // 4 (M) x 2 (N) waves
    const int l16 = lane & 15, half = lane >> 4;
    const int bm = blockIdx.y * 64, bn = blockIdx.x * 64;

    v8f acc0 = {}; v8f acc1 = {};

    for (int k0 = 0; k0 < K; k0 += 32) {
        {   // stage A: 8 floats/thread -> bf16
            int r = tid >> 2, c = (tid & 3) * 8;
            const float* src = A + (size_t)(bm + r) * lda + k0 + c;
            v4f f0 = *(const v4f*)src;
            v4f f1 = *(const v4f*)(src + 4);
            v8bf o;
            #pragma unroll
            for (int i = 0; i < 4; ++i) { o[i] = (__bf16)f0[i]; o[4 + i] = (__bf16)f1[i]; }
            *(v8bf*)&As[r][c] = o;
        }
        {   // stage B: one 16B copy/thread from packed buffer
            int n = tid >> 2, kk = (tid & 3) * 8;
            *(v8bf*)&Bs[n][kk] = *(const v8bf*)(Bp + (size_t)(bn + n) * K + k0 + kk);
        }
        __syncthreads();
        v16bf af = afrag_from(&As[wm * 16 + l16][0], 0, half);
        v16bf b0 = bfrag_from(&Bs[wn * 32 + l16][0],      0, half);
        v16bf b1 = bfrag_from(&Bs[wn * 32 + 16 + l16][0], 0, half);
        acc0 = __builtin_amdgcn_wmma_f32_16x16x32_bf16(false, af, false, b0, (short)0, acc0, false, false);
        acc1 = __builtin_amdgcn_wmma_f32_16x16x32_bf16(false, af, false, b1, (short)0, acc1, false, false);
        __syncthreads();
    }
    #pragma unroll
    for (int nt = 0; nt < 2; ++nt) {
        const v8f& a = nt ? acc1 : acc0;
        int col = bn + wn * 32 + nt * 16 + l16;
        float bv = bias ? bias[col] : 0.0f;
        #pragma unroll
        for (int i = 0; i < 8; ++i) {
            int row = bm + wm * 16 + half * 8 + i;
            C[(size_t)row * ldc + col] = a[i] + bv;
        }
    }
}

// ============== ef GEMM: C[E,1024] = h1[E,256] @ Wee + bee ==============
// B panel first 32 of 64 cols (x 256 K) preloaded once; A fragments loaded straight from global.
__global__ __launch_bounds__(256) void ef_gemm(
    const __bf16* __restrict__ h1,   // E x 256 row-major bf16
    const __bf16* __restrict__ Bp,   // 1024 x 256 packed [n][k]
    const float*  __restrict__ bias, // 1024
    float*        __restrict__ C)    // E x 1024
{
    __shared__ alignas(128) __bf16 Bs[64][256];  // 32 KB
    const int tid = threadIdx.x;
    const int bm = blockIdx.y * 64, bn = blockIdx.x * 64;
    {   // preload Bs rows 0..31 only: 8192 bf16 = 1024 x 16B, 4 per thread
        const v8bf* src = (const v8bf*)(Bp + (size_t)bn * 256);
        v8bf* dst = (v8bf*)&Bs[0][0];
        #pragma unroll
        for (int i = 0; i < 4; ++i) dst[tid + 256 * i] = src[tid + 256 * i];
    }
    __syncthreads();

    const int lane = tid & 31, wave = tid >> 5;
    const int wm = wave & 3, wn = wave >> 2;
    const int l16 = lane & 15, half = lane >> 4;
    const __bf16* arow = h1 + (size_t)(bm + wm * 16 + l16) * 256;
    const __bf16* brow0 = &Bs[wn * 32 + l16][0];
    const __bf16* brow1 = &Bs[wn * 32 + 16 + l16][0];

    v8f acc0 = {}; v8f acc1 = {};
    #pragma unroll
    for (int k0 = 0; k0 < 256; k0 += 32) {
        v16bf af = afrag_from(arow, k0, half);
        v16bf b0 = bfrag_from(brow0, k0, half);
        v16bf b1 = bfrag_from(brow1, k0, half);
        acc0 = __builtin_amdgcn_wmma_f32_16x16x32_bf16(false, af, false, b0, (short)0, acc0, false, false);
        acc1 = __builtin_amdgcn_wmma_f32_16x16x32_bf16(false, af, false, b1, (short)0, acc1, false, false);
    }
    #pragma unroll
    for (int nt = 0; nt < 2; ++nt) {
        const v8f& a = nt ? acc1 : acc0;
        int col = bn + wn * 32 + nt * 16 + l16;
        float bv = bias[col];
        #pragma unroll
        for (int i = 0; i < 8; ++i) {
            int row = bm + wm * 16 + half * 8 + i;
            C[(size_t)row * 1024 + col] = a[i] + bv;
        }
    }
}

// ============== fused per-target gate + aggregate ==============
// agg[t,col] = (1/191) * sum_{s != t} gelu( h1[e(s,t)] @ Wg[:,col] + bg[col] ) * x[s,col]
// Diagonal row s==t loads dummy edge 0 and is masked in the epilogue.
__global__ __launch_bounds__(256) void gate_agg_wmma(
    const __bf16* __restrict__ h1,   // E x 256 bf16
    const __bf16* __restrict__ Bp,   // Wg packed [n][k]: 1024 x 256 bf16
    const float*  __restrict__ bg,   // 1024
    const float*  __restrict__ x,    // 192 x 1024 current feats
    float*        __restrict__ agg)  // 192 x 1024
{
    const int t  = blockIdx.y;
    const int bn = blockIdx.x * 64;
    __shared__ alignas(128) __bf16 Bs[64][256];  // 32 KB
    __shared__ alignas(128) float  xs[64][64];   // 16 KB
    __shared__ float red2[16][64];               // 4 KB

    const int tid  = threadIdx.x;
    const int lane = tid & 31, wave = tid >> 5;
    const int wm = wave & 3, wn = wave >> 2;
    const int l16 = lane & 15, half = lane >> 4;

    {   // preload Wg panel rows 0..31 only: 1024 x 16B, 4 per thread
        const v8bf* src = (const v8bf*)(Bp + (size_t)bn * 256);
        v8bf* dst = (v8bf*)&Bs[0][0];
        #pragma unroll
        for (int i = 0; i < 4; ++i) dst[tid + 256 * i] = src[tid + 256 * i];
    }
    const __bf16* brow0 = &Bs[wn * 32 + l16][0];
    const __bf16* brow1 = &Bs[wn * 32 + 16 + l16][0];

    float part0 = 0.0f, part1 = 0.0f;

    for (int chunk = 0; chunk < 3; ++chunk) {
        {   // stage x tile (64 rows x 64 cols fp32), coalesced
            int r = tid >> 2, c4 = (tid & 3) * 16;
            const float* xp = x + (size_t)(chunk * 64 + r) * 1024 + bn + c4;
            #pragma unroll
            for (int i = 0; i < 4; ++i)
                *(v4f*)&xs[r][c4 + 4 * i] = *(const v4f*)(xp + 4 * i);
        }
        __syncthreads();   // covers Bs on first iteration too

        // per-lane A row: gather h1 row for edge (s -> t); dummy e=0 when s==t (masked later)
        int s_lane = chunk * 64 + wm * 16 + l16;
        int et = (t < s_lane) ? t : t - 1;
        size_t e = (s_lane == t) ? 0 : (size_t)s_lane * 191 + et;
        const __bf16* arow = h1 + e * 256;

        v8f acc0 = {}; v8f acc1 = {};
        #pragma unroll
        for (int k0 = 0; k0 < 256; k0 += 32) {
            v16bf af = afrag_from(arow, k0, half);
            v16bf b0 = bfrag_from(brow0, k0, half);
            v16bf b1 = bfrag_from(brow1, k0, half);
            acc0 = __builtin_amdgcn_wmma_f32_16x16x32_bf16(false, af, false, b0, (short)0, acc0, false, false);
            acc1 = __builtin_amdgcn_wmma_f32_16x16x32_bf16(false, af, false, b1, (short)0, acc1, false, false);
        }
        // epilogue: bias + gelu, multiply by x, mask diagonal, reduce rows
        #pragma unroll
        for (int nt = 0; nt < 2; ++nt) {
            const v8f& a = nt ? acc1 : acc0;
            int c64 = wn * 32 + nt * 16 + l16;
            float bv = bg[bn + c64];
            float p = 0.0f;
            #pragma unroll
            for (int i = 0; i < 8; ++i) {
                int rloc = wm * 16 + half * 8 + i;
                int s = chunk * 64 + rloc;
                float g = gelu_f(a[i] + bv);
                if (s != t) p += g * xs[rloc][c64];
            }
            if (nt) part1 += p; else part0 += p;
        }
        __syncthreads();   // protect xs restage
    }
    // deterministic cross-wave reduction
    red2[wave * 2 + half][wn * 32 + l16]      = part0;
    red2[wave * 2 + half][wn * 32 + 16 + l16] = part1;
    __syncthreads();
    if (tid < 64) {
        int c = tid, wnc = c >> 5;
        float s = 0.0f;
        #pragma unroll
        for (int wmi = 0; wmi < 4; ++wmi)
            #pragma unroll
            for (int hf = 0; hf < 2; ++hf)
                s += red2[((wnc * 4 + wmi) * 2) + hf][c];
        agg[(size_t)t * 1024 + bn + c] = s * (1.0f / 191.0f);
    }
}

// ============== small helper kernels ==============
__global__ __launch_bounds__(256) void vecmat(
    const float* __restrict__ v, const float* __restrict__ Wm,
    const float* __restrict__ badd, float* __restrict__ out, int K, int Nn)
{
    int n = blockIdx.x * 256 + threadIdx.x;
    if (n >= Nn) return;
    float s = badd ? badd[n] : 0.0f;
    for (int k = 0; k < K; ++k) s += v[k] * Wm[(size_t)k * Nn + n];
    out[n] = s;
}

// h1[e,:] = bf16(gelu(Af[src] + Bf[tgt] + b1)); 8 edges/block, 8 cols/lane
__global__ __launch_bounds__(256) void build_h1(
    const float* __restrict__ Af, const float* __restrict__ Bf,
    const float* __restrict__ b1, __bf16* __restrict__ h1)
{
    int e = blockIdx.x * 8 + (threadIdx.x >> 5);
    int c = (threadIdx.x & 31) * 8;
    int s = e / 191, r = e % 191;
    int t = (r < s) ? r : r + 1;
    const float* af = Af + s * 256 + c;
    const float* bf = Bf + t * 256 + c;
    const float* bb = b1 + c;
    v8bf o;
    #pragma unroll
    for (int i = 0; i < 8; ++i)
        o[i] = (__bf16)gelu_f(af[i] + bf[i] + bb[i]);
    *(v8bf*)(h1 + (size_t)e * 256 + c) = o;
}

__global__ __launch_bounds__(256) void concat2(
    const float* __restrict__ x, const float* __restrict__ agg, float* __restrict__ comb)
{
    int idx = blockIdx.x * 256 + threadIdx.x;     // 192*2048 total
    int r = idx >> 11, c = idx & 2047;
    comb[idx] = (c < 1024) ? x[r * 1024 + c] : agg[r * 1024 + (c - 1024)];
}

__global__ __launch_bounds__(256) void ln_gelu(
    const float* __restrict__ pre, const float* __restrict__ g,
    const float* __restrict__ b, float* __restrict__ out)
{
    int r = blockIdx.x;
    __shared__ float s1[256], s2[256];
    float lsum = 0.0f, lsq = 0.0f;
    for (int c = threadIdx.x; c < 1024; c += 256) {
        float v = pre[(size_t)r * 1024 + c]; lsum += v; lsq += v * v;
    }
    s1[threadIdx.x] = lsum; s2[threadIdx.x] = lsq;
    __syncthreads();
    for (int o = 128; o > 0; o >>= 1) {
        if (threadIdx.x < o) { s1[threadIdx.x] += s1[threadIdx.x + o]; s2[threadIdx.x] += s2[threadIdx.x + o]; }
        __syncthreads();
    }
    float mu  = s1[0] * (1.0f / 1024.0f);
    float var = s2[0] * (1.0f / 1024.0f) - mu * mu;
    float inv = rsqrtf(var + 1e-5f);
    for (int c = threadIdx.x; c < 1024; c += 256) {
        float v = (pre[(size_t)r * 1024 + c] - mu) * inv * g[c] + b[c];
        out[(size_t)r * 1024 + c] = gelu_f(v);
    }
}

__global__ __launch_bounds__(256) void head_kernel(
    const float* __restrict__ xf, const float* __restrict__ w1, const float* __restrict__ b1,
    const float* __restrict__ w2, const float* __restrict__ b2, float* __restrict__ gemb)
{
    __shared__ float pooled[1024];
    __shared__ float hid[512];
    int tid = threadIdx.x;
    for (int c = tid; c < 1024; c += 256) {
        float s = 0.0f;
        for (int r = 0; r < 192; ++r) s += xf[(size_t)r * 1024 + c];
        pooled[c] = s * (1.0f / 192.0f);
    }
    __syncthreads();
    for (int j = tid; j < 512; j += 256) {
        float s = b1[j];
        for (int k = 0; k < 1024; ++k) s += pooled[k] * w1[(size_t)k * 512 + j];
        hid[j] = gelu_f(s);
    }
    __syncthreads();
    for (int n = tid; n < 1024; n += 256) {
        float s = b2[n];
        for (int k = 0; k < 512; ++k) s += hid[k] * w2[(size_t)k * 1024 + n];
        gemb[n] = s;
    }
}

// ============== launch ==============
extern "C" void kernel_launch(void* const* d_in, const int* in_sizes, int n_in,
                              void* d_out, int out_size, void* d_ws, size_t ws_size,
                              hipStream_t stream)
{
    (void)in_sizes; (void)n_in; (void)out_size; (void)ws_size;
    const float* node_emb   = (const float*)d_in[0];
    const float* node_enc_w = (const float*)d_in[1];
    const float* node_enc_b = (const float*)d_in[2];
    const float* ep_w1      = (const float*)d_in[3];
    const float* ep_b1      = (const float*)d_in[4];
    const float* ep_w2      = (const float*)d_in[5];
    const float* ep_b2      = (const float*)d_in[6];
    const float* ee_w       = (const float*)d_in[7];
    const float* ee_b       = (const float*)d_in[8];
    const float* nm_w       = (const float*)d_in[9];
    const float* nm_b       = (const float*)d_in[10];
    const float* ln_g       = (const float*)d_in[11];
    const float* ln_b       = (const float*)d_in[12];
    const float* em_w       = (const float*)d_in[13];
    const float* em_b       = (const float*)d_in[14];
    const float* gp_w1      = (const float*)d_in[15];
    const float* gp_b1      = (const float*)d_in[16];
    const float* gp_w2      = (const float*)d_in[17];
    const float* gp_b2      = (const float*)d_in[18];

    float* out    = (float*)d_out;
    float* out_x  = out;
    float* out_ef = out + NNODE * DDIM;
    float* out_ge = out + NNODE * DDIM + (size_t)NEDGE * DDIM;

    char* wp = (char*)d_ws;
    auto alloc = [&](size_t bytes) -> char* {
        char* p = wp; wp += (bytes + 255) & ~(size_t)255; return p;
    };
    float*  xA   = (float*)alloc((size_t)NNODE * DDIM * 4);
    float*  xB   = (float*)alloc((size_t)NNODE * DDIM * 4);
    float*  Af   = (float*)alloc((size_t)NNODE * EDIM * 4);
    float*  Bf   = (float*)alloc((size_t)NNODE * EDIM * 4);
    float*  Wee  = (float*)alloc((size_t)EDIM * DDIM * 4);
    float*  bee  = (float*)alloc((size_t)DDIM * 4);
    float*  Wg   = (float*)alloc((size_t)3 * EDIM * DDIM * 4);
    float*  bg   = (float*)alloc((size_t)3 * DDIM * 4);
    __bf16* h1   = (__bf16*)alloc((size_t)NEDGE * EDIM * 2);
    float*  agg  = (float*)alloc((size_t)NNODE * DDIM * 4);
    float*  comb = (float*)alloc((size_t)NNODE * 2 * DDIM * 4);
    float*  pre  = (float*)alloc((size_t)NNODE * DDIM * 4);
    __bf16* Bpk  = (__bf16*)alloc((size_t)2 * DDIM * DDIM * 2); // shared 4MB pack buffer

    dim3 blk(256);
    // x0 = node_emb @ node_enc_w + b
    pack_b<<<dim3(DDIM/64, DDIM/64), blk, 0, stream>>>(node_enc_w, DDIM, Bpk, DDIM);
    gemm_wmma<<<dim3(DDIM/64, NNODE/64), blk, 0, stream>>>(node_emb, DDIM, Bpk, node_enc_b, xA, DDIM, DDIM);
    // Af = x0 @ ep_w1[:1024], Bf = x0 @ ep_w1[1024:]
    pack_b<<<dim3(EDIM/64, DDIM/64), blk, 0, stream>>>(ep_w1, EDIM, Bpk, DDIM);
    gemm_wmma<<<dim3(EDIM/64, NNODE/64), blk, 0, stream>>>(xA, DDIM, Bpk, nullptr, Af, EDIM, DDIM);
    pack_b<<<dim3(EDIM/64, DDIM/64), blk, 0, stream>>>(ep_w1 + (size_t)DDIM * EDIM, EDIM, Bpk, DDIM);
    gemm_wmma<<<dim3(EDIM/64, NNODE/64), blk, 0, stream>>>(xA, DDIM, Bpk, nullptr, Bf, EDIM, DDIM);
    // Wee = ep_w2 @ ee_w ; bee = ep_b2 @ ee_w + ee_b
    pack_b<<<dim3(DDIM/64, EDIM/64), blk, 0, stream>>>(ee_w, DDIM, Bpk, EDIM);
    gemm_wmma<<<dim3(DDIM/64, EDIM/64), blk, 0, stream>>>(ep_w2, EDIM, Bpk, nullptr, Wee, DDIM, EDIM);
    vecmat<<<dim3(DDIM/256), blk, 0, stream>>>(ep_b2, ee_w, ee_b, bee, EDIM, DDIM);
    // Wg[l] = Wee @ em_w[l] ; bg[l] = bee @ em_w[l] + em_b[l]
    for (int l = 0; l < 3; ++l) {
        pack_b<<<dim3(DDIM/64, DDIM/64), blk, 0, stream>>>(em_w + (size_t)l * DDIM * DDIM, DDIM, Bpk, DDIM);
        gemm_wmma<<<dim3(DDIM/64, EDIM/64), blk, 0, stream>>>(Wee, DDIM, Bpk, nullptr,
                                                              Wg + (size_t)l * EDIM * DDIM, DDIM, DDIM);
        vecmat<<<dim3(DDIM/256), blk, 0, stream>>>(bee, em_w + (size_t)l * DDIM * DDIM,
                                                   em_b + (size_t)l * DDIM, bg + (size_t)l * DDIM, DDIM, DDIM);
    }
    // h1 = gelu(Af[src] + Bf[tgt] + ep_b1)  (E x 256 bf16)
    build_h1<<<dim3(NEDGE / 8), blk, 0, stream>>>(Af, Bf, ep_b1, h1);
    // ef = h1 @ Wee + bee  (heavy, output region)
    pack_b<<<dim3(DDIM/64, EDIM/64), blk, 0, stream>>>(Wee, DDIM, Bpk, EDIM);
    ef_gemm<<<dim3(DDIM/64, NEDGE/64), blk, 0, stream>>>(h1, Bpk, bee, out_ef);

    const float* xcur = xA;
    for (int l = 0; l < 3; ++l) {
        // pack Wg[l] then fused gate+aggregate
        pack_b<<<dim3(DDIM/64, EDIM/64), blk, 0, stream>>>(Wg + (size_t)l * EDIM * DDIM, DDIM, Bpk, EDIM);
        gate_agg_wmma<<<dim3(DDIM/64, NNODE), blk, 0, stream>>>(h1, Bpk, bg + (size_t)l * DDIM, xcur, agg);
        // node update
        concat2<<<dim3(NNODE * 2 * DDIM / 256), blk, 0, stream>>>(xcur, agg, comb);
        pack_b<<<dim3(DDIM/64, 2*DDIM/64), blk, 0, stream>>>(nm_w + (size_t)l * 2 * DDIM * DDIM, DDIM, Bpk, 2 * DDIM);
        gemm_wmma<<<dim3(DDIM/64, NNODE/64), blk, 0, stream>>>(comb, 2 * DDIM, Bpk,
                                                               nm_b + (size_t)l * DDIM, pre, DDIM, 2 * DDIM);
        float* xn = (l == 2) ? out_x : ((xcur == xA) ? xB : xA);
        ln_gelu<<<dim3(NNODE), blk, 0, stream>>>(pre, ln_g + (size_t)l * DDIM, ln_b + (size_t)l * DDIM, xn);
        xcur = xn;
    }
    head_kernel<<<dim3(1), blk, 0, stream>>>(out_x, gp_w1, gp_b1, gp_w2, gp_b2, out_ge);
}
